// ScaledDotProductAttention_36558761624004
// MI455X (gfx1250) — matrix operand order repaired and checked by reading, not compiled, not run on hardware
//
#include <hip/hip_runtime.h>

typedef __attribute__((ext_vector_type(16))) __bf16 v16bf;
typedef __attribute__((ext_vector_type(8)))  __bf16 v8bf;
typedef __attribute__((ext_vector_type(8)))  float  v8f;
typedef __attribute__((ext_vector_type(4)))  float  v4f;

namespace {
constexpr int   B_      = 2;
constexpr int   H_      = 16;
constexpr int   S_      = 2048;
constexpr int   D_      = 64;
constexpr float kScale  = 0.125f;    // 1/sqrt(64)
constexpr float kNegInf = -1.0e9f;

__device__ __forceinline__ float rowMax16(float x) {
  x = fmaxf(x, __shfl_xor(x, 1, 32));
  x = fmaxf(x, __shfl_xor(x, 2, 32));
  x = fmaxf(x, __shfl_xor(x, 4, 32));
  x = fmaxf(x, __shfl_xor(x, 8, 32));
  return x;
}
__device__ __forceinline__ float rowSum16(float x) {
  x += __shfl_xor(x, 1, 32);
  x += __shfl_xor(x, 2, 32);
  x += __shfl_xor(x, 4, 32);
  x += __shfl_xor(x, 8, 32);
  return x;
}
} // namespace

__global__ __launch_bounds__(256)
void fmha_fwd_causal_kernel(const float* __restrict__ Q, const float* __restrict__ K,
                            const float* __restrict__ V, float* __restrict__ ctx,
                            float* __restrict__ attn) {
  // Per-wave staging for a 16x32 P block: bf16 copy feeds the P@V A-fragment,
  // fp32 copy feeds wide coalesced non-temporal global stores of the attention matrix.
  __shared__ __bf16 stageB[8][16 * 32];
  __shared__ float  stageF[8][16 * 32];

  const int lane = threadIdx.x & 31;
  const int wv   = threadIdx.x >> 5;
  const int hi   = lane >> 4;                    // half-wave id
  const int lo   = lane & 15;
  const int bh   = blockIdx.y;
  const int q0   = blockIdx.x * 128 + wv * 16;   // this wave's 16 query rows
  const int diag = q0 >> 4;                      // last key tile with live entries

  const float* Qb = Q + (size_t)bh * S_ * D_;
  const float* Kb = K + (size_t)bh * S_ * D_;
  const float* Vb = V + (size_t)bh * S_ * D_;
  float* Cb = ctx  + (size_t)bh * S_ * D_;
  float* Ab = attn + (size_t)bh * S_ * S_;

  // ---- Q stripe -> two bf16 A fragments (16x32 each), 1/sqrt(d) folded in.
  // A k order used here (contiguous, 16x32): M = lane&15; K = 16*hi + e
  v16bf qa0, qa1;
  {
    const float4* q4 = (const float4*)(Qb + (size_t)(q0 + lo) * D_);
    #pragma unroll
    for (int f = 0; f < 2; ++f) {
      const int d0 = f * 32;
      float t[16];
      *(float4*)&t[0]  = q4[((d0 + 16 * hi) >> 2) + 0];
      *(float4*)&t[4]  = q4[((d0 + 16 * hi) >> 2) + 1];
      *(float4*)&t[8]  = q4[((d0 + 16 * hi) >> 2) + 2];
      *(float4*)&t[12] = q4[((d0 + 16 * hi) >> 2) + 3];
      v16bf a;
      #pragma unroll
      for (int e = 0; e < 16; ++e) a[e] = (__bf16)(t[e] * kScale);
      if (f == 0) qa0 = a; else qa1 = a;
    }
  }

  // B k order used here (contiguous, 32x16): lane = N + 16*(K>>4), e = K & 15.
  // Scores B = K^T chunk: per lane, 16 contiguous d values starting at d0 + 16*hi.
  // Regular-temporal loads on purpose: K/V (32 MiB total) should stay L2-resident
  // across the three passes; only the 536 MB write-once output stream is NT.
  auto loadK = [&](int kb, int d0) {
    const float4* k4 = (const float4*)(Kb + (size_t)(kb + lo) * D_);
    const int base = (d0 + 16 * hi) >> 2;
    float t[16];
    *(float4*)&t[0]  = k4[base + 0];
    *(float4*)&t[4]  = k4[base + 1];
    *(float4*)&t[8]  = k4[base + 2];
    *(float4*)&t[12] = k4[base + 3];
    v16bf b;
    #pragma unroll
    for (int e = 0; e < 16; ++e) b[e] = (__bf16)t[e];
    return b;
  };

  auto scoreTile = [&](int kt) {
    v16bf b0 = loadK(kt * 16, 0);
    v16bf b1 = loadK(kt * 16, 32);
    v8f s = {};
    s = __builtin_amdgcn_wmma_f32_16x16x32_bf16(false, qa0, false, b0, (short)0, s, false, false);
    s = __builtin_amdgcn_wmma_f32_16x16x32_bf16(false, qa1, false, b1, (short)0, s, false, false);
    return s;
  };

  // ============ Pass 1: per-lane partial row max; single cross-lane reduce ============
  float m[8];
  #pragma unroll
  for (int v = 0; v < 8; ++v) m[v] = -3.0e38f;

  for (int kt = 0; kt <= diag; ++kt) {
    v8f s = scoreTile(kt);
    const int kcol = kt * 16 + lo;
    #pragma unroll
    for (int v = 0; v < 8; ++v) {
      const int qrow = q0 + v + 8 * hi;           // C/D layout: M = v + 8*hi, N = lo
      m[v] = fmaxf(m[v], (kcol <= qrow) ? s[v] : kNegInf);
    }
  }
  #pragma unroll
  for (int v = 0; v < 8; ++v) m[v] = rowMax16(m[v]);

  // ============ Pass 2: per-lane partial row sum of exp(s-m); single reduce ============
  float l[8];
  #pragma unroll
  for (int v = 0; v < 8; ++v) l[v] = 0.0f;

  for (int kt = 0; kt <= diag; ++kt) {
    v8f s = scoreTile(kt);
    const int kcol = kt * 16 + lo;
    #pragma unroll
    for (int v = 0; v < 8; ++v) {
      const int qrow = q0 + v + 8 * hi;
      const float sv = (kcol <= qrow) ? s[v] : kNegInf;
      l[v] += __expf(sv - m[v]);
    }
  }
  float invl[8];
  #pragma unroll
  for (int v = 0; v < 8; ++v) invl[v] = 1.0f / rowSum16(l[v]);

  // ============ Pass 3: emit normalized P (wide NT stores) and O = P @ V ============
  const int npairs = (diag >> 1) + 1;             // 32-key steps
  v8f o0 = {}, o1 = {}, o2 = {}, o3 = {};

  for (int j = 0; j < npairs; ++j) {
    const int ktA = 2 * j;
    const int ktB = 2 * j + 1;
    const bool liveB = (ktB <= diag);

    // --- tile A probabilities -> stage cols [0,16)
    {
      v8f s = scoreTile(ktA);
      const int kcol = ktA * 16 + lo;
      #pragma unroll
      for (int v = 0; v < 8; ++v) {
        const int qrow = q0 + v + 8 * hi;
        const int r    = v + 8 * hi;
        float p = (kcol <= qrow) ? __expf(s[v] - m[v]) * invl[v] : 0.0f;
        stageB[wv][r * 32 + lo] = (__bf16)p;
        stageF[wv][r * 32 + lo] = p;
      }
    }
    // --- tile B probabilities -> stage cols [16,32)
    if (liveB) {
      v8f s = scoreTile(ktB);
      const int kcol = ktB * 16 + lo;
      #pragma unroll
      for (int v = 0; v < 8; ++v) {
        const int qrow = q0 + v + 8 * hi;
        const int r    = v + 8 * hi;
        float p = (kcol <= qrow) ? __expf(s[v] - m[v]) * invl[v] : 0.0f;
        stageB[wv][r * 32 + 16 + lo] = (__bf16)p;
        stageF[wv][r * 32 + 16 + lo] = p;
      }
    } else {
      #pragma unroll
      for (int v = 0; v < 8; ++v) {
        const int r = v + 8 * hi;
        stageB[wv][r * 32 + 16 + lo] = (__bf16)0.0f;
        stageF[wv][r * 32 + 16 + lo] = 0.0f;
      }
    }
    asm volatile("s_wait_dscnt 0" ::: "memory");

    // --- coalesced NT fp32 store of the 16x32 block: 2 lanes per row, 4 vec4 each
    {
      const int row = lane >> 1;
      const int c4  = (lane & 1) * 4;
      v4f* dst = (v4f*)(Ab + (size_t)(q0 + row) * S_ + j * 32);
      const v4f* src = (const v4f*)&stageF[wv][row * 32];
      #pragma unroll
      for (int i = 0; i < 4; ++i) __builtin_nontemporal_store(src[c4 + i], &dst[c4 + i]);
    }

    // --- P block -> A fragment: two contiguous 16B LDS loads per lane
    v16bf ap;
    {
      v8bf a0 = *(const v8bf*)&stageB[wv][lo * 32 + 16 * hi];
      v8bf a1 = *(const v8bf*)&stageB[wv][lo * 32 + 16 * hi + 8];
      #pragma unroll
      for (int e = 0; e < 8; ++e) { ap[e] = a0[e]; ap[8 + e] = a1[e]; }
    }

    // --- V block (32 keys x 64 d) -> 4 B fragments; this lane's keys: ktA*16 + 16*hi + e
    const bool vlive = (ktA + hi) <= diag;        // guards OOB + masked half
    #pragma unroll
    for (int c = 0; c < 4; ++c) {
      v16bf bv;
      #pragma unroll
      for (int e = 0; e < 16; ++e) {
        float x = 0.0f;
        if (vlive) x = Vb[(size_t)(ktA * 16 + 16 * hi + e) * D_ + c * 16 + lo];
        bv[e] = (__bf16)x;
      }
      v8f acc = (c == 0) ? o0 : (c == 1) ? o1 : (c == 2) ? o2 : o3;
      acc = __builtin_amdgcn_wmma_f32_16x16x32_bf16(false, ap, false, bv, (short)0, acc, false, false);
      if (c == 0) o0 = acc; else if (c == 1) o1 = acc; else if (c == 2) o2 = acc; else o3 = acc;
    }
  }

  // ---- write context (16 x 64 fp32), non-temporal (write-once, never re-read)
  #pragma unroll
  for (int v = 0; v < 8; ++v) {
    float* row = Cb + (size_t)(q0 + v + 8 * hi) * D_;
    __builtin_nontemporal_store(o0[v], row +  0 + lo);
    __builtin_nontemporal_store(o1[v], row + 16 + lo);
    __builtin_nontemporal_store(o2[v], row + 32 + lo);
    __builtin_nontemporal_store(o3[v], row + 48 + lo);
  }

  // ---- NT zero-fill of the fully-masked columns beyond the emitted pair blocks
  const int cstart4 = npairs * 8;                 // (npairs*32)/4
  const v4f zero4 = {0.f, 0.f, 0.f, 0.f};
  for (int r = 0; r < 16; ++r) {
    v4f* row4 = (v4f*)(Ab + (size_t)(q0 + r) * S_);
    for (int c = cstart4 + lane; c < S_ / 4; c += 32)
      __builtin_nontemporal_store(zero4, &row4[c]);
  }
}

extern "C" void kernel_launch(void* const* d_in, const int* in_sizes, int n_in,
                              void* d_out, int out_size, void* d_ws, size_t ws_size,
                              hipStream_t stream) {
  (void)in_sizes; (void)n_in; (void)d_ws; (void)ws_size; (void)out_size;
  const float* Q = (const float*)d_in[0];
  const float* K = (const float*)d_in[1];
  const float* V = (const float*)d_in[2];
  // d_in[3] is the boolean causal mask; the kernel applies causal masking analytically.
  float* ctx  = (float*)d_out;                                   // [B,H,S,D]
  float* attn = (float*)d_out + (size_t)B_ * H_ * S_ * D_;       // [B,H,S,S]

  dim3 grid(S_ / 128, B_ * H_);
  dim3 block(256);
  fmha_fwd_causal_kernel<<<grid, block, 0, stream>>>(Q, K, V, ctx, attn);
}
